// Capsule_26276609916989
// MI455X (gfx1250) — matrix operand order repaired and checked by reading, not compiled, not run on hardware
//
#include <hip/hip_runtime.h>
#include <hip/hip_bf16.h>
#include <math.h>

// ---------------------------------------------------------------------------
// Capsule routing for MI455X (gfx1250, wave32, WMMA).
//   B=32, In=2048, Din=256, N=16 capsules, D=32, 3 routing iterations.
// Phase 1: u_hat = u_vecs @ W via split-precision bf16 WMMA (AhBh+AlBh+AhBl),
//          written directly in (B, N, In, D) layout into workspace.
// Phase 2: 3x { softmax-over-capsules + target-modulated contraction + squash,
//              then logit update }, all streaming the L2-resident u_hat.
// ---------------------------------------------------------------------------

typedef __attribute__((ext_vector_type(16))) __bf16 v16bf;
typedef __attribute__((ext_vector_type(8)))  float  v8f;
typedef __attribute__((ext_vector_type(4)))  float  v4f;

#define CAP_B   32
#define CAP_IN  2048
#define CAP_DIN 256
#define CAP_N   16
#define CAP_D   32
#define CAP_NF  (CAP_N * CAP_D)   // 512
#define CAP_EPS 1e-7f

// LDS W-strip: 64 columns x 256 K, bf16 hi+lo, padded row stride (16B aligned)
#define WLDS_STRIDE 272           // elements per column row (256 + 16 pad)

// ===========================================================================
// Kernel 1: u_hat GEMM.  grid = (NF/64, M/128), block = 256 (8 waves).
// Each wave: one 16-row M strip x the block's 64-wide N strip.
// A fragment (16x32 bf16): lane L<16 -> row L, K = 0..15 of the step;
//                          lane L>=16 -> row L-16, K = 16..31.
// B fragment (32x16 bf16): lane L<16 -> col L,  K = 0..15;
//                          lane L>=16 -> col L-16, K = 16..31.
// C (16x16 f32, 8 VGPR): vgpr r, lane L -> row r + 8*(L>=16), col L&15.
// ===========================================================================
__global__ __launch_bounds__(256)
void capsule_gemm_uhat(const float* __restrict__ A,   // (B*In, 256) = u_vecs
                       const float* __restrict__ W,   // (256, 512)
                       float* __restrict__ uh)        // (B, N, In, D)
{
    extern __shared__ __align__(16) char smem_raw[];
    __bf16* wh = (__bf16*)smem_raw;                 // [64][WLDS_STRIDE]
    __bf16* wl = wh + 64 * WLDS_STRIDE;

    const int n0 = blockIdx.x * 64;                 // N-strip base (in NF)
    const int m_blk = blockIdx.y * 128;             // M-strip base

    // ---- Stage W strip to LDS as split bf16 (transposed: [n][k]) ----
    for (int idx = threadIdx.x; idx < 256 * 64; idx += 256) {
        const int k = idx >> 6;
        const int nn = idx & 63;
        const float w = W[(size_t)k * CAP_NF + n0 + nn];
        const __bf16 h = (__bf16)w;
        const __bf16 l = (__bf16)(w - (float)h);
        wh[nn * WLDS_STRIDE + k] = h;
        wl[nn * WLDS_STRIDE + k] = l;
    }
    __syncthreads();

    const int lane  = threadIdx.x & 31;
    const int wave  = threadIdx.x >> 5;
    const int khalf = lane >> 4;                    // 0: lanes 0-15, 1: 16-31
    const int lrow  = lane & 15;

    const int m0w = m_blk + wave * 16;              // this wave's 16 rows
    const float* arow = A + (size_t)(m0w + lrow) * CAP_DIN;

    v8f acc[4] = {};                                // 4 N-tiles of 16

    #pragma unroll
    for (int ks = 0; ks < CAP_DIN / 32; ++ks) {     // 8 K-steps of 32
        const int k0 = ks * 32;

        // ---- A fragment (split hi/lo), 4x contiguous 16B loads per lane ----
        const int kA = k0 + khalf * 16;
        float af[16];
        *(v4f*)(af + 0)  = *(const v4f*)(arow + kA);
        *(v4f*)(af + 4)  = *(const v4f*)(arow + kA + 4);
        *(v4f*)(af + 8)  = *(const v4f*)(arow + kA + 8);
        *(v4f*)(af + 12) = *(const v4f*)(arow + kA + 12);

        v16bf ah, al;
        #pragma unroll
        for (int j = 0; j < 16; ++j) {
            const __bf16 h = (__bf16)af[j];
            ah[j] = h;
            al[j] = (__bf16)(af[j] - (float)h);
        }

        // ---- 4 N-tiles: B frags from LDS, 3 split-term WMMAs each ----
        #pragma unroll
        for (int nt = 0; nt < 4; ++nt) {
            const int coff = (nt * 16 + lrow) * WLDS_STRIDE + k0 + khalf * 16;
            const v16bf bh = *(const v16bf*)(wh + coff);
            const v16bf bl = *(const v16bf*)(wl + coff);

            acc[nt] = __builtin_amdgcn_wmma_f32_16x16x32_bf16(
                false, ah, false, bh, (short)0, acc[nt], false, false);
            acc[nt] = __builtin_amdgcn_wmma_f32_16x16x32_bf16(
                false, al, false, bh, (short)0, acc[nt], false, false);
            acc[nt] = __builtin_amdgcn_wmma_f32_16x16x32_bf16(
                false, ah, false, bl, (short)0, acc[nt], false, false);
        }
    }

    // ---- Store C into (B, N, In, D) layout ----
    const int bb = m0w >> 11;                       // batch (128-aligned tiles
    const int i_base = m0w & 2047;                  //   never straddle b)
    #pragma unroll
    for (int nt = 0; nt < 4; ++nt) {
        const int f = n0 + nt * 16 + lrow;          // column in NF
        const int ncap = f >> 5;
        const int d    = f & 31;
        float* dst = uh + ((((size_t)bb * CAP_N + ncap) * CAP_IN) + i_base) * CAP_D + d;
        #pragma unroll
        for (int r = 0; r < 8; ++r) {
            const int mi = r + 8 * khalf;           // row within 16-tile
            dst[(size_t)mi * CAP_D] = acc[nt][r];
        }
    }
}

// ===========================================================================
// Kernel 2: softmax over capsules + coupling + contraction + squash.
// grid = B*N workgroups, block = 256. Logits buffer layout: [b][i][16].
// ===========================================================================
__global__ __launch_bounds__(256)
void capsule_route_outputs(const float* __restrict__ uh,     // (B,N,In,D)
                           const float* __restrict__ target, // (B,N,In)
                           const float* __restrict__ blog,   // (B,In,N)
                           float* __restrict__ out,          // (B,N,D)
                           int first)
{
    const int bn = blockIdx.x;
    const int b  = bn >> 4;
    const int n  = bn & 15;
    const float* uhbn = uh + (size_t)bn * CAP_IN * CAP_D;
    const float* tg   = target + (size_t)bn * CAP_IN;

    float acc[CAP_D];
    #pragma unroll
    for (int d = 0; d < CAP_D; ++d) acc[d] = 0.f;

    for (int i = threadIdx.x; i < CAP_IN; i += 256) {
        float c;
        if (first) {
            c = 1.0f / 16.0f;                        // softmax(0) over 16 caps
        } else {
            const float* brow = blog + ((size_t)b * CAP_IN + i) * CAP_N;
            float v[16];
            *(v4f*)(v + 0)  = *(const v4f*)(brow + 0);
            *(v4f*)(v + 4)  = *(const v4f*)(brow + 4);
            *(v4f*)(v + 8)  = *(const v4f*)(brow + 8);
            *(v4f*)(v + 12) = *(const v4f*)(brow + 12);
            float m = v[0];
            #pragma unroll
            for (int j = 1; j < 16; ++j) m = fmaxf(m, v[j]);
            float s = 0.f;
            #pragma unroll
            for (int j = 0; j < 16; ++j) s += __expf(v[j] - m);
            c = __expf(v[n] - m) / s;
        }
        const float t1 = tg[i] * c;
        const v4f* urow = (const v4f*)(uhbn + (size_t)i * CAP_D);
        #pragma unroll
        for (int q = 0; q < CAP_D / 4; ++q) {
            const v4f u = urow[q];
            acc[q * 4 + 0] = fmaf(t1, u.x, acc[q * 4 + 0]);
            acc[q * 4 + 1] = fmaf(t1, u.y, acc[q * 4 + 1]);
            acc[q * 4 + 2] = fmaf(t1, u.z, acc[q * 4 + 2]);
            acc[q * 4 + 3] = fmaf(t1, u.w, acc[q * 4 + 3]);
        }
    }

    // Wave32 reduction, then cross-wave via LDS.
    #pragma unroll
    for (int d = 0; d < CAP_D; ++d) {
        #pragma unroll
        for (int off = 16; off > 0; off >>= 1)
            acc[d] += __shfl_down(acc[d], off, 32);
    }
    __shared__ float wsum[8][CAP_D];
    const int lane = threadIdx.x & 31;
    const int wave = threadIdx.x >> 5;
    if (lane == 0) {
        #pragma unroll
        for (int d = 0; d < CAP_D; ++d) wsum[wave][d] = acc[d];
    }
    __syncthreads();
    if (wave == 0) {                                 // lane == d (32 lanes)
        float s = 0.f;
        #pragma unroll
        for (int w = 0; w < 8; ++w) s += wsum[w][lane];
        float sq = s * s;
        #pragma unroll
        for (int off = 16; off > 0; off >>= 1)
            sq += __shfl_xor(sq, off, 32);
        out[(size_t)bn * CAP_D + lane] = s / sqrtf(sq + CAP_EPS);  // squash
    }
}

// ===========================================================================
// Kernel 3: logit update  b[b,i,n] = sum_d out[b,n,d] * uh[b,n,i,d].
// One thread per (b,n,i).  grid = B*N*In/256.
// ===========================================================================
__global__ __launch_bounds__(256)
void capsule_route_logits(const float* __restrict__ uh,   // (B,N,In,D)
                          const float* __restrict__ out,  // (B,N,D)
                          float* __restrict__ blog)       // (B,In,N)
{
    const size_t t = (size_t)blockIdx.x * blockDim.x + threadIdx.x;
    const int i  = (int)(t & (CAP_IN - 1));
    const int bn = (int)(t >> 11);
    const int b  = bn >> 4;
    const int n  = bn & 15;

    const v4f* urow = (const v4f*)(uh + ((size_t)bn * CAP_IN + i) * CAP_D);
    const v4f* orow = (const v4f*)(out + (size_t)bn * CAP_D);  // wave-uniform
    float s = 0.f;
    #pragma unroll
    for (int q = 0; q < CAP_D / 4; ++q) {
        const v4f u = urow[q];
        const v4f o = orow[q];
        s = fmaf(u.x, o.x, s);
        s = fmaf(u.y, o.y, s);
        s = fmaf(u.z, o.z, s);
        s = fmaf(u.w, o.w, s);
    }
    blog[((size_t)b * CAP_IN + i) * CAP_N + n] = s;
}

// ===========================================================================
extern "C" void kernel_launch(void* const* d_in, const int* in_sizes, int n_in,
                              void* d_out, int out_size, void* d_ws, size_t ws_size,
                              hipStream_t stream)
{
    (void)in_sizes; (void)n_in; (void)out_size; (void)ws_size;

    const float* u_vecs = (const float*)d_in[0];   // (32, 2048, 256)
    const float* target = (const float*)d_in[1];   // (32, 16, 2048)
    const float* W      = (const float*)d_in[2];   // (256, 512)
    float* out = (float*)d_out;                    // (32, 16, 32)

    float* uh   = (float*)d_ws;                                    // 128 MiB
    float* blog = uh + (size_t)CAP_B * CAP_N * CAP_IN * CAP_D;     // 4 MiB

    // Phase 1: split-bf16 WMMA GEMM (M=65536, K=256, N=512).
    const dim3 gg(CAP_NF / 64, (CAP_B * CAP_IN) / 128);
    const size_t lds_bytes = (size_t)2 * 64 * WLDS_STRIDE * 2;     // hi+lo bf16
    capsule_gemm_uhat<<<gg, 256, lds_bytes, stream>>>(u_vecs, W, uh);

    // Phase 2: 3 routing iterations (first uses softmax(0) = 1/16).
    for (int it = 0; it < 3; ++it) {
        capsule_route_outputs<<<CAP_B * CAP_N, 256, 0, stream>>>(
            uh, target, blog, out, it == 0 ? 1 : 0);
        if (it < 2) {
            capsule_route_logits<<<(CAP_B * CAP_N * CAP_IN) / 256, 256, 0, stream>>>(
                uh, out, blog);
        }
    }
}
